// Attention_84722524880900
// MI455X (gfx1250) — matrix operand order repaired and checked by reading, not compiled, not run on hardware
//
#include <hip/hip_runtime.h>
#include <hip/hip_bf16.h>

typedef __bf16 bf16_t;
typedef __attribute__((ext_vector_type(16))) __bf16 v16bf;
typedef __attribute__((ext_vector_type(8)))  float  v8f;
typedef __attribute__((ext_vector_type(4)))  unsigned int u32x4;
typedef __attribute__((ext_vector_type(8)))  int  i32x8;
typedef __attribute__((ext_vector_type(4)))  int  i32x4;

#define NSEQ   1024
#define DIMSZ  1024
#define BATCH  8
#define NHEAD  16
#define DHEAD  64
#define QKVLD  3072          // row stride of qkv (3*inner), elements
#define MASKVAL (-3.4028235e38f)

// Tensor Data Mover availability (this toolchain: 6-arg builtin). Host pass
// and toolchains without the builtin fall back to manual staging.
#if defined(__HIP_DEVICE_COMPILE__) && defined(__has_builtin)
#  if __has_builtin(__builtin_amdgcn_tensor_load_to_lds) && \
      __has_builtin(__builtin_amdgcn_s_wait_tensorcnt)
#    define USE_TDM 1
#  endif
#endif
#ifndef USE_TDM
#  define USE_TDM 0
#endif

// Assemble a v16bf fragment from two 16-byte chunks.
union frag_u { uint4 u[2]; v16bf v; };

// A-fragment (16x32 bf16, row-major LDS tile with 32-elem rows):
// lane row = l16; elements = [base..base+15], base=hlf*16.
__device__ __forceinline__ v16bf load_a_frag(const bf16_t* rowp, int hlf) {
  frag_u f;
  f.u[0] = *(const uint4*)(rowp + hlf * 16);
  f.u[1] = *(const uint4*)(rowp + 8 + hlf * 16);
  return f.v;
}
// B-fragment from K-contiguous storage: 16 contiguous elements at kbase=hlf*16.
__device__ __forceinline__ v16bf load_b_frag(const bf16_t* colp, int hlf) {
  frag_u f;
  const uint4* p = (const uint4*)(colp + hlf * 16);
  f.u[0] = p[0];
  f.u[1] = p[1];
  return f.v;
}

// ---------------------------------------------------------------------------
// f32 -> bf16 cast
// ---------------------------------------------------------------------------
__global__ void cast_f32_bf16(const float* __restrict__ in,
                              bf16_t* __restrict__ out, int n) {
  int i = blockIdx.x * blockDim.x + threadIdx.x;
  if (i < n) out[i] = (bf16_t)in[i];
}

// ---------------------------------------------------------------------------
// Generic bf16 WMMA GEMM: C[M,N] = A[M,K] * B[K,N] (+bias), A/B row-major.
// Block = 128 threads = 4 waves; tile 64x64. B tile stored K-contiguous
// (transposed) in LDS so B-fragments are b128 loads.
// ---------------------------------------------------------------------------
template<bool OUT_BF16, bool ADD_BIAS>
__global__ void __launch_bounds__(128)
gemm_bf16_wmma(const bf16_t* __restrict__ A, const bf16_t* __restrict__ B,
               float* __restrict__ Cf, bf16_t* __restrict__ Cb,
               const float* __restrict__ bias, int M, int N, int K) {
  __shared__ __align__(16) bf16_t As[64 * 32];    // [row][k]
  __shared__ __align__(16) bf16_t BsT[64 * 32];   // [col][k]
  const int tid  = threadIdx.x;
  const int wave = tid >> 5, lane = tid & 31;
  const int hlf  = lane >> 4, l16 = lane & 15;
  const int m0 = blockIdx.y * 64, n0 = blockIdx.x * 64;

  v8f acc[4] = {};
  for (int k0 = 0; k0 < K; k0 += 32) {
    // stage A tile (64 rows x 32 k), b128 copies
    for (int idx = tid; idx < 64 * 4; idx += 128) {
      int r = idx >> 2, c = (idx & 3) * 8;
      *(uint4*)&As[r * 32 + c] =
          *(const uint4*)(A + (size_t)(m0 + r) * K + k0 + c);
    }
    // stage B tile (32 k x 64 cols) transposed -> BsT[col][k]
    for (int idx = tid; idx < 32 * 8; idx += 128) {
      int r = idx >> 3, c = (idx & 7) * 8;
      bf16_t tmp[8];
      *(uint4*)tmp = *(const uint4*)(B + (size_t)(k0 + r) * N + n0 + c);
      #pragma unroll
      for (int t = 0; t < 8; ++t) BsT[(c + t) * 32 + r] = tmp[t];
    }
    if (k0 + 32 < K)
      __builtin_prefetch(A + (size_t)(m0 + (tid >> 1)) * K + k0 + 32, 0, 0);
    __syncthreads();

    v16bf a = load_a_frag(&As[(wave * 16 + l16) * 32], hlf);
    #pragma unroll
    for (int nt = 0; nt < 4; ++nt) {
      v16bf bb = load_b_frag(&BsT[(nt * 16 + l16) * 32], hlf);
      acc[nt] = __builtin_amdgcn_wmma_f32_16x16x32_bf16(
          false, a, false, bb, (short)0, acc[nt], false, false);
    }
    __syncthreads();
  }

  #pragma unroll
  for (int nt = 0; nt < 4; ++nt)
    #pragma unroll
    for (int r = 0; r < 8; ++r) {
      int row = m0 + wave * 16 + hlf * 8 + r;
      int col = n0 + nt * 16 + l16;
      float v = acc[nt][r];
      if (ADD_BIAS) v += bias[col];
      if (OUT_BF16) Cb[(size_t)row * N + col] = (bf16_t)v;
      else          Cf[(size_t)row * N + col] = v;
    }
}

// ---------------------------------------------------------------------------
// Pooled logits: attn_raw[b,i,j] = SCALE_ * sum_k q_[b,i,k] * k_[b,j,k]
// ---------------------------------------------------------------------------
__global__ void __launch_bounds__(128)
pooled_logits(const bf16_t* __restrict__ qkv, float* __restrict__ attn) {
  __shared__ __align__(16) bf16_t As[64 * 32];    // [qrow][k]
  __shared__ __align__(16) bf16_t BsT[64 * 32];   // [jcol][k]
  const int tid  = threadIdx.x;
  const int wave = tid >> 5, lane = tid & 31;
  const int hlf  = lane >> 4, l16 = lane & 15;
  const int n0 = blockIdx.x * 64, m0 = blockIdx.y * 64, b = blockIdx.z;

  const bf16_t* qm = qkv + (size_t)b * NSEQ * QKVLD;   // q_ slice
  const bf16_t* km = qm + DIMSZ;                       // k_ slice

  v8f acc[4] = {};
  for (int k0 = 0; k0 < DIMSZ; k0 += 32) {
    for (int idx = tid; idx < 64 * 4; idx += 128) {
      int r = idx >> 2, c = (idx & 3) * 8;
      *(uint4*)&As[r * 32 + c] =
          *(const uint4*)(qm + (size_t)(m0 + r) * QKVLD + k0 + c);
    }
    for (int idx = tid; idx < 64 * 4; idx += 128) {
      int r = idx >> 2, c = (idx & 3) * 8;
      bf16_t tmp[8];
      *(uint4*)tmp = *(const uint4*)(km + (size_t)(n0 + r) * QKVLD + k0 + c);
      #pragma unroll
      for (int t = 0; t < 8; ++t) BsT[r * 32 + c + t] = tmp[t];
    }
    __syncthreads();

    v16bf a = load_a_frag(&As[(wave * 16 + l16) * 32], hlf);
    #pragma unroll
    for (int nt = 0; nt < 4; ++nt) {
      v16bf bb = load_b_frag(&BsT[(nt * 16 + l16) * 32], hlf);
      acc[nt] = __builtin_amdgcn_wmma_f32_16x16x32_bf16(
          false, a, false, bb, (short)0, acc[nt], false, false);
    }
    __syncthreads();
  }

  #pragma unroll
  for (int nt = 0; nt < 4; ++nt)
    #pragma unroll
    for (int r = 0; r < 8; ++r) {
      int row = m0 + wave * 16 + hlf * 8 + r;
      int col = n0 + nt * 16 + l16;
      attn[((size_t)b * NSEQ + row) * NSEQ + col] = acc[nt][r] * 0.03125f;
    }
}

// ---------------------------------------------------------------------------
// In-place row softmax, one block per row.
// ---------------------------------------------------------------------------
__global__ void __launch_bounds__(256)
softmax_rows(float* __restrict__ data, int ncols) {
  __shared__ float red[256];
  float* row = data + (size_t)blockIdx.x * ncols;
  const int tid = threadIdx.x;
  float mx = MASKVAL;
  for (int c = tid; c < ncols; c += 256) mx = fmaxf(mx, row[c]);
  red[tid] = mx; __syncthreads();
  for (int s = 128; s > 0; s >>= 1) {
    if (tid < s) red[tid] = fmaxf(red[tid], red[tid + s]);
    __syncthreads();
  }
  mx = red[0]; __syncthreads();
  float sum = 0.f;
  for (int c = tid; c < ncols; c += 256) {
    float e = __expf(row[c] - mx); row[c] = e; sum += e;
  }
  red[tid] = sum; __syncthreads();
  for (int s = 128; s > 0; s >>= 1) {
    if (tid < s) red[tid] += red[tid + s];
    __syncthreads();
  }
  float inv = 1.0f / red[0];
  for (int c = tid; c < ncols; c += 256) row[c] *= inv;
}

// ---------------------------------------------------------------------------
// Flash-style masked multi-head attention; one wave = 16 query rows.
// K tile staged via Tensor Data Mover (TDM) when available.
// Grid: (n/64, heads, batch), block 128.
// ---------------------------------------------------------------------------
__global__ void __launch_bounds__(128)
flash_mha(const bf16_t* __restrict__ qkv, const unsigned char* __restrict__ msk,
          bf16_t* __restrict__ ctx) {
  __shared__ __align__(16) bf16_t Ks[32 * 64];     // [key][dim] row-major
  __shared__ __align__(16) bf16_t VsT[64 * 32];    // [dim][key] (transposed)
  __shared__ __align__(16) bf16_t Ps[4][16 * 32];  // per-wave P tile
  __shared__ unsigned char Jm[32];

  const int tid  = threadIdx.x;
  const int wave = tid >> 5, lane = tid & 31;
  const int hlf  = lane >> 4, l16 = lane & 15;
  const int qb = blockIdx.x, h = blockIdx.y, b = blockIdx.z;

  const bf16_t* qbase = qkv + (size_t)b * NSEQ * QKVLD + (size_t)h * DHEAD;
  const bf16_t* kbase = qbase + DIMSZ;
  const bf16_t* vbase = qbase + 2 * DIMSZ;
  const int qrow0 = qb * 64 + wave * 16;

  // Q fragments (global b128 loads)
  v16bf aq[2];
  {
    const bf16_t* qrow = qbase + (size_t)(qrow0 + l16) * QKVLD;
    #pragma unroll
    for (int s = 0; s < 2; ++s) {
      frag_u f;
      f.u[0] = *(const uint4*)(qrow + s * 32 + hlf * 16);
      f.u[1] = *(const uint4*)(qrow + s * 32 + 8 + hlf * 16);
      aq[s] = f.v;
    }
  }

  float qmr[8], mrow[8], lrow[8];
  #pragma unroll
  for (int r = 0; r < 8; ++r) {
    int row = qrow0 + hlf * 8 + r;
    qmr[r]  = (row == 0) ? 1.f : (msk[(size_t)b * (NSEQ - 1) + row - 1] ? 1.f : 0.f);
    mrow[r] = MASKVAL;
    lrow[r] = 0.f;
  }
  v8f o[4] = {};

  for (int j0 = 0; j0 < NSEQ; j0 += 32) {
#if USE_TDM
    // K tile (32 keys x 64 dims, 2B elements, row stride QKVLD) via TDM.
    if (wave == 0) {
      unsigned long long ga =
          (unsigned long long)(uintptr_t)(kbase + (size_t)j0 * QKVLD);
      unsigned ldsa = (unsigned)(uintptr_t)&Ks[0];
      u32x4 g0 = (u32x4)0u;
      g0.x = 1u;                                        // count=1, user mode
      g0.y = ldsa;                                      // lds_addr (bytes)
      g0.z = (unsigned)(ga & 0xFFFFFFFFu);              // global_addr[31:0]
      g0.w = (unsigned)((ga >> 32) & 0x01FFFFFFu)       // global_addr[56:32]
           | (2u << 30);                                // type = 2 (image)
      i32x8 g1 = (i32x8)0;
      g1[0] = (int)(1u << 16);        // data_size = 2 bytes
      g1[1] = (int)(64u << 16);       // tensor_dim0 = 64 (bits 79:48, lo 16)
      g1[2] = (int)(32u << 16);       // tensor_dim1 = 32 (bits 111:80, lo 16)
      g1[3] = (int)(64u << 16);       // tile_dim0 = 64 (bits 127:112)
      g1[4] = 32;                     // tile_dim1 = 32 (bits 143:128)
      g1[5] = QKVLD;                  // tensor_dim0_stride (bits 191:160)
      i32x4 gz4 = (i32x4)0;
      i32x8 gz8 = (i32x8)0;
      __builtin_amdgcn_tensor_load_to_lds(g0, g1, gz4, gz4, gz8, 0);
    }
#else
    for (int idx = tid; idx < 32 * 8; idx += 128) {
      int r = idx >> 3, c = (idx & 7) * 8;
      *(uint4*)&Ks[r * 64 + c] =
          *(const uint4*)(kbase + (size_t)(j0 + r) * QKVLD + c);
    }
#endif
    // V tile transposed -> VsT[dim][key]
    for (int idx = tid; idx < 32 * 8; idx += 128) {
      int r = idx >> 3, c = (idx & 7) * 8;
      bf16_t tmp[8];
      *(uint4*)tmp = *(const uint4*)(vbase + (size_t)(j0 + r) * QKVLD + c);
      #pragma unroll
      for (int t = 0; t < 8; ++t) VsT[(c + t) * 32 + r] = tmp[t];
    }
    if (tid < 32) {
      int j = j0 + tid;
      Jm[tid] = (j == 0) ? (unsigned char)1 : msk[(size_t)b * (NSEQ - 1) + j - 1];
    }
    if (j0 + 32 < NSEQ)
      __builtin_prefetch(kbase + (size_t)(j0 + 32 + (tid & 31)) * QKVLD, 0, 0);
#if USE_TDM
    if (wave == 0) __builtin_amdgcn_s_wait_tensorcnt(0);
#endif
    __syncthreads();

    // S = Q K^T : B fragment is K-contiguous in Ks (row-major [key][dim])
    v8f sv[2];
    #pragma unroll
    for (int jt = 0; jt < 2; ++jt) {
      v8f s = {};
      #pragma unroll
      for (int ks = 0; ks < 2; ++ks) {
        frag_u f;
        const uint4* kp =
            (const uint4*)&Ks[(jt * 16 + l16) * 64 + ks * 32 + hlf * 16];
        f.u[0] = kp[0]; f.u[1] = kp[1];
        s = __builtin_amdgcn_wmma_f32_16x16x32_bf16(
            false, aq[ks], false, f.v, (short)0, s, false, false);
      }
      sv[jt] = s;
    }

    const float cm0 = Jm[l16]      ? 1.f : 0.f;
    const float cm1 = Jm[16 + l16] ? 1.f : 0.f;

    #pragma unroll
    for (int r = 0; r < 8; ++r) {
      float s0 = sv[0][r] * 0.125f;
      float s1 = sv[1][r] * 0.125f;
      if (!(qmr[r] != 0.f && cm0 != 0.f)) s0 = MASKVAL;
      if (!(qmr[r] != 0.f && cm1 != 0.f)) s1 = MASKVAL;
      float mx = fmaxf(s0, s1);
      #pragma unroll
      for (int sh = 1; sh < 16; sh <<= 1)
        mx = fmaxf(mx, __shfl_xor(mx, sh, 32));
      float mnew = fmaxf(mrow[r], mx);
      float corr = __expf(mrow[r] - mnew);
      mrow[r] = mnew;
      float p0 = __expf(s0 - mnew);
      float p1 = __expf(s1 - mnew);
      float ls = p0 + p1;
      #pragma unroll
      for (int sh = 1; sh < 16; sh <<= 1)
        ls += __shfl_xor(ls, sh, 32);
      lrow[r] = lrow[r] * corr + ls;
      #pragma unroll
      for (int nt = 0; nt < 4; ++nt) o[nt][r] *= corr;
      Ps[wave][(hlf * 8 + r) * 32 + l16]      = (bf16_t)p0;
      Ps[wave][(hlf * 8 + r) * 32 + 16 + l16] = (bf16_t)p1;
    }
    __syncthreads();

    // O += P V  (A-frag from Ps, B-frag K-contiguous in VsT)
    v16bf ap = load_a_frag(&Ps[wave][l16 * 32], hlf);
    #pragma unroll
    for (int nt = 0; nt < 4; ++nt) {
      v16bf bv = load_b_frag(&VsT[(nt * 16 + l16) * 32], hlf);
      o[nt] = __builtin_amdgcn_wmma_f32_16x16x32_bf16(
          false, ap, false, bv, (short)0, o[nt], false, false);
    }
    __syncthreads();
  }

  // normalize and write context (bf16, feeds the final projection GEMM)
  #pragma unroll
  for (int r = 0; r < 8; ++r) {
    float inv = 1.0f / lrow[r];
    int row = qrow0 + hlf * 8 + r;
    bf16_t* dst = ctx + ((size_t)b * NSEQ + row) * DIMSZ + (size_t)h * DHEAD;
    #pragma unroll
    for (int nt = 0; nt < 4; ++nt)
      dst[nt * 16 + l16] = (bf16_t)(o[nt][r] * inv);
  }
}

// ---------------------------------------------------------------------------
extern "C" void kernel_launch(void* const* d_in, const int* in_sizes, int n_in,
                              void* d_out, int out_size, void* d_ws, size_t ws_size,
                              hipStream_t stream) {
  const float*         x     = (const float*)d_in[0];
  const unsigned char* msk   = (const unsigned char*)d_in[1];
  const float*         w_qkv = (const float*)d_in[2];
  const float*         w_out = (const float*)d_in[3];
  const float*         b_out = (const float*)d_in[4];

  float* out  = (float*)d_out;                                   // [8,1024,1024]
  float* attn = out + (size_t)BATCH * NSEQ * DIMSZ;              // [8,1024,1024]

  char* ws = (char*)d_ws;
  bf16_t* x_bf    = (bf16_t*)(ws);                               // 16 MB
  bf16_t* wqkv_bf = (bf16_t*)(ws + (16ull << 20));               //  6 MB
  bf16_t* wout_bf = (bf16_t*)(ws + (22ull << 20));               //  2 MB
  bf16_t* qkv_bf  = (bf16_t*)(ws + (24ull << 20));               // 48 MB
  bf16_t* ctx_bf  = (bf16_t*)(ws + (72ull << 20));               // 16 MB

  const int n_x  = BATCH * NSEQ * DIMSZ;      // 8388608
  const int n_w1 = DIMSZ * 3 * DIMSZ;         // 3145728
  const int n_w2 = DIMSZ * DIMSZ;             // 1048576
  cast_f32_bf16<<<n_x  / 256, 256, 0, stream>>>(x,     x_bf,    n_x);
  cast_f32_bf16<<<n_w1 / 256, 256, 0, stream>>>(w_qkv, wqkv_bf, n_w1);
  cast_f32_bf16<<<n_w2 / 256, 256, 0, stream>>>(w_out, wout_bf, n_w2);

  // qkv = x @ w_qkv  (bf16 output)
  dim3 g1(3 * DIMSZ / 64, BATCH * NSEQ / 64);
  gemm_bf16_wmma<true, false><<<g1, 128, 0, stream>>>(
      x_bf, wqkv_bf, nullptr, qkv_bf, nullptr,
      BATCH * NSEQ, 3 * DIMSZ, DIMSZ);

  // pooled attention (no mask): raw logits then row softmax, in d_out
  pooled_logits<<<dim3(NSEQ / 64, NSEQ / 64, BATCH), 128, 0, stream>>>(qkv_bf, attn);
  softmax_rows<<<BATCH * NSEQ, 256, 0, stream>>>(attn, NSEQ);

  // masked multi-head attention -> context (bf16)
  flash_mha<<<dim3(NSEQ / 64, NHEAD, BATCH), 128, 0, stream>>>(qkv_bf, msk, ctx_bf);

  // out = ctx @ w_out + b_out  (f32 output)
  dim3 g2(DIMSZ / 64, BATCH * NSEQ / 64);
  gemm_bf16_wmma<false, true><<<g2, 128, 0, stream>>>(
      ctx_bf, wout_bf, out, nullptr, b_out,
      BATCH * NSEQ, DIMSZ, DIMSZ);
}
